// TutorialDecoder_52132313038890
// MI455X (gfx1250) — matrix operand order repaired and checked by reading, not compiled, not run on hardware
//
#include <hip/hip_runtime.h>
#include <hip/hip_bf16.h>
#include <math.h>

// ---------------------------------------------------------------------------
// TutorialDecoder on MI455X (gfx1250, wave32, WMMA 16x16x32 f16 -> f32 acc)
//
//  * All weights converted f32 -> f16 once (~86.5 MB) -> fully L2-resident
//    (192 MB L2); steady-state weight reads never touch HBM (23.3 TB/s only
//    sees the 2 MB/step output writes).
//  * M=16 equals one WMMA tile: each wave owns one 16x16 output tile and
//    runs a K-loop of v_wmma_f32_16x16x32_f16.
//  * All GEMM operands pre-converted to f16 so inner loops are pure
//    load_b128 + wmma (no cvt chains between WMMAs).
//  * att_in = [h | context] kept as one contiguous f16 buffer (stride 2048);
//    the LSTM cell writes h straight into it.
//  * 200 serial steps -> 7 small kernels/step on `stream` (graph-capturable).
// ---------------------------------------------------------------------------

typedef __attribute__((ext_vector_type(16))) _Float16 v16h;
typedef __attribute__((ext_vector_type(8)))  _Float16 v8h;
typedef __attribute__((ext_vector_type(8)))  float    v8f;

#define BB     16
#define HH     1024
#define II     1024
#define VV     32000
#define NSTEPS 200
#define GG     4096   // 4*HH
#define AK     2048   // HH + II (att_in row stride)

// ----------------------------- WMMA helpers --------------------------------

__device__ __forceinline__ v8f wmma_f16(v16h a, v16h b, v8f c) {
  return __builtin_amdgcn_wmma_f32_16x16x32_f16(
      /*neg_a=*/false, a, /*neg_b=*/false, b,
      /*c_mod=*/(short)0, c, /*reuse_a=*/false, /*reuse_b=*/false);
}

// A-fragment (16x32 f16, row-major activation).
// Lane l holds row m = l&15.  Lanes 0-15: K = k0+[0..15];
// lanes 16-31: K = k0+16+[0..15].  Two aligned 16B loads.
__device__ __forceinline__ v16h frag_a(const _Float16* row, int k0, int lane) {
  const int kb = k0 + ((lane & 16) ? 16 : 0);
  v8h lo = *reinterpret_cast<const v8h*>(row + kb);
  v8h hi = *reinterpret_cast<const v8h*>(row + kb + 8);
  v16h a;
#pragma unroll
  for (int e = 0; e < 8; ++e) { a[e] = lo[e]; a[e + 8] = hi[e]; }
  return a;
}

// B-fragment (32x16 = K x N) from row-major weight W[N][ldk]:
// B[k][n] = W[n0+n][k0+k].  Lane l holds column n = l&15; lanes 0-15 hold
// K = k0+[0..15], lanes 16-31 hold K = k0+16+[0..15]: one aligned 32B load.
__device__ __forceinline__ v16h frag_b(const _Float16* W, int ldk, int n0, int k0, int lane) {
  const _Float16* p = W + (size_t)(n0 + (lane & 15)) * ldk
                        + (size_t)(k0 + ((lane & 16) ? 16 : 0));
  return *reinterpret_cast<const v16h*>(p);
}

// ----------------------------- setup kernels -------------------------------

__global__ void cvt_f32_to_f16_kernel(const float* __restrict__ src,
                                      _Float16* __restrict__ dst, int n) {
  for (int i = blockIdx.x * blockDim.x + threadIdx.x; i < n;
       i += gridDim.x * blockDim.x)
    dst[i] = (_Float16)src[i];
}

// Zero c / h-half of att_in, fill context-half of att_in, reset tok.
__global__ void init_state_kernel(const float* __restrict__ ctx,
                                  float* __restrict__ c,
                                  _Float16* __restrict__ attin,
                                  int* __restrict__ tok) {
  int i = blockIdx.x * blockDim.x + threadIdx.x;       // 16384
  int b = i >> 10, j = i & (HH - 1);
  c[i] = 0.0f;
  attin[(size_t)b * AK + j]      = (_Float16)0.0f;              // h part
  attin[(size_t)b * AK + HH + j] = (_Float16)ctx[(size_t)b * II + j];
  if (i < BB) tok[i] = 0;
}

// Gather x = emb[tok] and convert to f16 once per step.
__global__ void embed_kernel(const float* __restrict__ emb,
                             const int* __restrict__ tok,
                             _Float16* __restrict__ x) {
  int i = blockIdx.x * blockDim.x + threadIdx.x;       // 16384
  int b = i >> 10, j = i & (HH - 1);
  x[i] = (_Float16)emb[(size_t)tok[b] * HH + j];
}

// ----------------------------- LSTM gates GEMM -----------------------------
// gates[16,4096] = x @ w_ih^T + h @ w_hh^T (+biases); 256 tiles -> 64 x 128.
__global__ __launch_bounds__(128)
void gates_kernel(const _Float16* __restrict__ x,
                  const _Float16* __restrict__ wih,
                  const _Float16* __restrict__ whh,
                  const float* __restrict__ bih,
                  const float* __restrict__ bhh,
                  const _Float16* __restrict__ attin,   // h rows, stride AK
                  float* __restrict__ gates) {
  const int lane = threadIdx.x & 31;
  const int wave = threadIdx.x >> 5;
  const int n0   = (blockIdx.x * 4 + wave) * 16;
  const int m    = lane & 15;

  const _Float16* xrow = x + (size_t)m * HH;
  const _Float16* hrow = attin + (size_t)m * AK;

  v8f acc = {};
#pragma unroll 4
  for (int k0 = 0; k0 < HH; k0 += 32) {
    v16h ax = frag_a(xrow, k0, lane);
    v16h bx = frag_b(wih, HH, n0, k0, lane);
    acc = wmma_f16(ax, bx, acc);
    v16h ah = frag_a(hrow, k0, lane);
    v16h bh = frag_b(whh, HH, n0, k0, lane);
    acc = wmma_f16(ah, bh, acc);
  }

  const int nn   = n0 + (lane & 15);
  const int mb   = (lane & 16) ? 8 : 0;
  const float bv = bih[nn] + bhh[nn];
#pragma unroll
  for (int e = 0; e < 8; ++e)
    gates[(size_t)(mb + e) * GG + nn] = acc[e] + bv;
}

// ----------------------------- LSTM cell -----------------------------------
__device__ __forceinline__ float sigf(float v) { return 1.0f / (1.0f + __expf(-v)); }

__global__ __launch_bounds__(256)
void cell_kernel(const float* __restrict__ gates,
                 float* __restrict__ c,
                 _Float16* __restrict__ attin) {
  int idx = blockIdx.x * blockDim.x + threadIdx.x;     // 16384
  int b = idx >> 10, j = idx & (HH - 1);
  const float* g = gates + (size_t)b * GG;
  float ig = sigf(g[j]);
  float fg = sigf(g[HH + j]);
  float gg = tanhf(g[2 * HH + j]);
  float og = sigf(g[3 * HH + j]);
  float cn = fg * c[idx] + ig * gg;
  c[idx] = cn;
  attin[(size_t)b * AK + j] = (_Float16)(og * tanhf(cn));  // h half of att_in
}

// ----------------------------- attention GEMM ------------------------------
// logits[16,1024] = att_in @ w_att^T ; w_att [1024,2048]; 64 tiles -> 16 x 128.
__global__ __launch_bounds__(128)
void att_kernel(const _Float16* __restrict__ attin,
                const _Float16* __restrict__ watt,
                const float* __restrict__ batt,
                float* __restrict__ logits) {
  const int lane = threadIdx.x & 31;
  const int wave = threadIdx.x >> 5;
  const int n0   = (blockIdx.x * 4 + wave) * 16;
  const int m    = lane & 15;

  const _Float16* arow = attin + (size_t)m * AK;

  v8f acc = {};
#pragma unroll 4
  for (int k0 = 0; k0 < AK; k0 += 32) {
    v16h a = frag_a(arow, k0, lane);
    v16h b = frag_b(watt, AK, n0, k0, lane);
    acc = wmma_f16(a, b, acc);
  }

  const int nn = n0 + (lane & 15);
  const int mb = (lane & 16) ? 8 : 0;
  const float bv = batt[nn];
#pragma unroll
  for (int e = 0; e < 8; ++e)
    logits[(size_t)(mb + e) * HH + nn] = acc[e] + bv;
}

// ----------------------------- softmax (per row) ---------------------------
__global__ __launch_bounds__(256)
void softmax_kernel(const float* __restrict__ logits, _Float16* __restrict__ aw) {
  __shared__ float red[256];
  const int row = blockIdx.x, t = threadIdx.x;
  const float* x = logits + (size_t)row * HH;

  float mx = -INFINITY;
  for (int j = t; j < HH; j += 256) mx = fmaxf(mx, x[j]);
  red[t] = mx; __syncthreads();
  for (int s = 128; s > 0; s >>= 1) {
    if (t < s) red[t] = fmaxf(red[t], red[t + s]);
    __syncthreads();
  }
  mx = red[0]; __syncthreads();

  float e[4];
  float sum = 0.0f;
  {
    int i = 0;
    for (int j = t; j < HH; j += 256, ++i) { e[i] = __expf(x[j] - mx); sum += e[i]; }
  }
  red[t] = sum; __syncthreads();
  for (int s = 128; s > 0; s >>= 1) {
    if (t < s) red[t] += red[t + s];
    __syncthreads();
  }
  const float inv = 1.0f / red[0];
  {
    int i = 0;
    for (int j = t; j < HH; j += 256, ++i)
      aw[(size_t)row * HH + j] = (_Float16)(e[i] * inv);
  }
}

// ----------------------------- output GEMM ---------------------------------
// out[b, step, :] = aw @ w_out^T + b_out ; 2000 tiles -> 500 x 128.
__global__ __launch_bounds__(128)
void out_kernel(const _Float16* __restrict__ aw,
                const _Float16* __restrict__ wout,
                const float* __restrict__ bout,
                float* __restrict__ out, int step) {
  const int lane = threadIdx.x & 31;
  const int wave = threadIdx.x >> 5;
  const int n0   = (blockIdx.x * 4 + wave) * 16;
  const int m    = lane & 15;

  const _Float16* arow = aw + (size_t)m * HH;

  v8f acc = {};
#pragma unroll 4
  for (int k0 = 0; k0 < HH; k0 += 32) {
    v16h a = frag_a(arow, k0, lane);
    v16h b = frag_b(wout, HH, n0, k0, lane);
    acc = wmma_f16(a, b, acc);
  }

  const int nn = n0 + (lane & 15);
  const int mb = (lane & 16) ? 8 : 0;
  const float bv = bout[nn];
#pragma unroll
  for (int e = 0; e < 8; ++e)
    out[((size_t)(mb + e) * NSTEPS + step) * VV + nn] = acc[e] + bv;
}

// ----------------------------- argmax (per row) ----------------------------
__global__ __launch_bounds__(256)
void argmax_kernel(const float* __restrict__ out, int* __restrict__ tok, int step) {
  __shared__ float rv[256];
  __shared__ int   ri[256];
  const int row = blockIdx.x, t = threadIdx.x;
  const float* x = out + ((size_t)row * NSTEPS + step) * VV;

  float bv = -INFINITY; int bi = 0x7fffffff;
  for (int j = t; j < VV; j += 256) {
    float v = x[j];
    if (v > bv) { bv = v; bi = j; }   // strictly > keeps first occurrence
  }
  rv[t] = bv; ri[t] = bi; __syncthreads();
  for (int s = 128; s > 0; s >>= 1) {
    if (t < s) {
      if (rv[t + s] > rv[t] || (rv[t + s] == rv[t] && ri[t + s] < ri[t])) {
        rv[t] = rv[t + s]; ri[t] = ri[t + s];
      }
    }
    __syncthreads();
  }
  if (t == 0) tok[row] = ri[0];
}

// ----------------------------- host driver ---------------------------------

extern "C" void kernel_launch(void* const* d_in, const int* in_sizes, int n_in,
                              void* d_out, int out_size, void* d_ws, size_t ws_size,
                              hipStream_t stream) {
  (void)in_sizes; (void)n_in; (void)out_size;

  const float* ctx  = (const float*)d_in[0];   // [16,1024]
  const float* emb  = (const float*)d_in[1];   // [32000,1024]
  const float* wih  = (const float*)d_in[2];   // [4096,1024]
  const float* whh  = (const float*)d_in[3];   // [4096,1024]
  const float* bih  = (const float*)d_in[4];   // [4096]
  const float* bhh  = (const float*)d_in[5];   // [4096]
  const float* watt = (const float*)d_in[6];   // [1024,2048]
  const float* batt = (const float*)d_in[7];   // [1024]
  const float* wout = (const float*)d_in[8];   // [32000,1024]
  const float* bout = (const float*)d_in[9];   // [32000]
  float* out = (float*)d_out;                  // [16,200,32000]

  // ---- workspace carve-up (256B aligned offsets) ----
  char* ws = (char*)d_ws;
  size_t off = 0;
  auto carve = [&](size_t bytes) -> void* {
    void* p = ws + off;
    off = (off + bytes + 255) & ~(size_t)255;
    return p;
  };
  const int nWih  = GG * HH;          // 4,194,304
  const int nWhh  = GG * HH;
  const int nWatt = HH * AK;          // 2,097,152
  const int nWout = VV * HH;          // 32,768,000

  _Float16* wih_h   = (_Float16*)carve((size_t)nWih  * 2);
  _Float16* whh_h   = (_Float16*)carve((size_t)nWhh  * 2);
  _Float16* watt_h  = (_Float16*)carve((size_t)nWatt * 2);
  _Float16* wout_h  = (_Float16*)carve((size_t)nWout * 2);
  _Float16* attin_h = (_Float16*)carve((size_t)BB * AK * 2);  // [h | ctx]
  _Float16* x_h     = (_Float16*)carve((size_t)BB * HH * 2);
  _Float16* aw_h    = (_Float16*)carve((size_t)BB * HH * 2);
  float*    c_f     = (float*)   carve((size_t)BB * HH * 4);
  float*    gates   = (float*)   carve((size_t)BB * GG * 4);
  float*    logits  = (float*)   carve((size_t)BB * HH * 4);
  int*      tok     = (int*)     carve((size_t)BB * 4);
  if (off > ws_size) return;  // workspace too small; nothing safe to do

  // ---- one-time (per call) f32 -> f16 weight conversion ----
  cvt_f32_to_f16_kernel<<<512,  256, 0, stream>>>(wih,  wih_h,  nWih);
  cvt_f32_to_f16_kernel<<<512,  256, 0, stream>>>(whh,  whh_h,  nWhh);
  cvt_f32_to_f16_kernel<<<512,  256, 0, stream>>>(watt, watt_h, nWatt);
  cvt_f32_to_f16_kernel<<<2048, 256, 0, stream>>>(wout, wout_h, nWout);

  init_state_kernel<<<64, 256, 0, stream>>>(ctx, c_f, attin_h, tok);

  // ---- 200 serial decode steps ----
  for (int s = 0; s < NSTEPS; ++s) {
    embed_kernel  <<< 64, 256, 0, stream>>>(emb, tok, x_h);
    gates_kernel  <<< 64, 128, 0, stream>>>(x_h, wih_h, whh_h, bih, bhh, attin_h, gates);
    cell_kernel   <<< 64, 256, 0, stream>>>(gates, c_f, attin_h);
    att_kernel    <<< 16, 128, 0, stream>>>(attin_h, watt_h, batt, logits);
    softmax_kernel<<< 16, 256, 0, stream>>>(logits, aw_h);
    out_kernel    <<<500, 128, 0, stream>>>(aw_h, wout_h, bout, out, s);
    argmax_kernel <<< 16, 256, 0, stream>>>(out, tok, s);
  }
}
